// DGRACL_58523224375313
// MI455X (gfx1250) — matrix operand order repaired and checked by reading, not compiled, not run on hardware
//
#include <hip/hip_runtime.h>
#include <hip/hip_bf16.h>
#include <float.h>

// Problem constants (match reference)
#define N_Q   1024
#define M_P   100000
#define DIM   256
#define KSEL  7
#define EPSF  1e-8f
#define ALPHA 0.6f
#define BETA  0.4f

#define NSLICE 50
#define SLICE_M (M_P / NSLICE)          // 2000
#define TILES_PER_SLICE (SLICE_M / 16)  // 125
#define QTILES (N_Q / 16)               // 64

typedef __attribute__((ext_vector_type(16))) __bf16 bf16x16;
typedef __attribute__((ext_vector_type(8)))  __bf16 bf16x8;
typedef __attribute__((ext_vector_type(8)))  float  f32x8;

// ---------------------------------------------------------------------------
// Row L2-normalize: out_bf16[row] = row / max(||row||, eps) (bf16), optional 1/||row||
// ---------------------------------------------------------------------------
__global__ __launch_bounds__(256) void norm_rows_kernel(const float* __restrict__ x,
                                                        __bf16* __restrict__ xn,
                                                        float* __restrict__ xinv) {
    const int row = blockIdx.x;
    const int t = threadIdx.x;
    __shared__ float red[256];
    float v = x[(size_t)row * DIM + t];
    red[t] = v * v;
    __syncthreads();
    for (int s = 128; s > 0; s >>= 1) {
        if (t < s) red[t] += red[t + s];
        __syncthreads();
    }
    float inv = 1.0f / fmaxf(sqrtf(red[0]), EPSF);
    xn[(size_t)row * DIM + t] = (__bf16)(v * inv);
    if (t == 0 && xinv) xinv[row] = inv;
}

// ---------------------------------------------------------------------------
// Similarity (bf16 WMMA) + time-decay weight + per-slice top-7 per query.
// One wave handles a 16-query tile across one M-slice of 2000 pool rows.
// Selection-phase only: exact fp32 rescore happens in fuse_kernel, so the
// hardware transcendental (__expf -> v_exp_f32) is used here.
// ---------------------------------------------------------------------------
__global__ __launch_bounds__(32) void sim_topk_kernel(const __bf16* __restrict__ qn,
                                                      const __bf16* __restrict__ pn,
                                                      const float* __restrict__ qt,
                                                      const float* __restrict__ pt,
                                                      const float* __restrict__ lam_p,
                                                      int* __restrict__ g_top) {
    const int lane  = threadIdx.x;
    const int half  = lane >> 4;   // 0: lanes 0-15, 1: lanes 16-31
    const int l16   = lane & 15;
    const int qtile = blockIdx.x;
    const int slice = blockIdx.y;
    const int q0    = qtile * 16;
    const int pbase = slice * SLICE_M;
    const float nlam = -lam_p[0];

    // --- A fragments (16x256 bf16 of normalized queries), ISA 16-bit A layout:
    // lane (m=l16, half): elems 0..7 -> K = c*32 + half*8 + j ; 8..15 -> K = c*32+16+half*8+j
    bf16x16 afrag[8];
    {
        const __bf16* qrow = qn + (size_t)(q0 + l16) * DIM;
#pragma unroll
        for (int c = 0; c < 8; ++c) {
            bf16x8 lo = *(const bf16x8*)(qrow + c * 32 + half * 8);
            bf16x8 hi = *(const bf16x8*)(qrow + c * 32 + 16 + half * 8);
#pragma unroll
            for (int t = 0; t < 8; ++t) { afrag[c][t] = lo[t]; afrag[c][8 + t] = hi[t]; }
        }
    }

    // C-tile row m = j + 8*half for VGPR j -> query times per lane
    float tq8[8];
#pragma unroll
    for (int j = 0; j < 8; ++j) tq8[j] = qt[q0 + j + 8 * half];

    float tv[KSEL]; int ti[KSEL];
#pragma unroll
    for (int j = 0; j < KSEL; ++j) { tv[j] = -FLT_MAX; ti[j] = 0; }

    __shared__ float sc[16][17];

    for (int tile = 0; tile < TILES_PER_SLICE; ++tile) {
        const int pcol = pbase + tile * 16 + l16;
        const __bf16* prow = pn + (size_t)pcol * DIM;
        if (tile + 1 < TILES_PER_SLICE)
            __builtin_prefetch(pn + (size_t)(pcol + 16) * DIM, 0, 1);  // global_prefetch_b8

        f32x8 acc = {};
#pragma unroll
        for (int c = 0; c < 8; ++c) {
            // B layout (32x16 bf16): lane holds column n=l16, elems 0..7 -> K = c*32 + half*8 + j ; 8..15 -> K = c*32+16+half*8+j
            bf16x8 lo = *(const bf16x8*)(prow + c * 32 + half * 8);
            bf16x8 hi = *(const bf16x8*)(prow + c * 32 + 16 + half * 8);
            bf16x16 bfrag;
#pragma unroll
            for (int t = 0; t < 8; ++t) { bfrag[t] = lo[t]; bfrag[8 + t] = hi[t]; }
            acc = __builtin_amdgcn_wmma_f32_16x16x32_bf16(false, afrag[c], false, bfrag,
                                                          (short)0, acc, false, false);
        }

        // apply time-decay weight (hw transcendental); stage tile in LDS to transpose
        const float tp = pt[pcol];
#pragma unroll
        for (int j = 0; j < 8; ++j) {
            float tw = __expf(nlam * fabsf(tq8[j] - tp));
            sc[j + 8 * half][l16] = acc[j] * tw;
        }
        __syncthreads();

        // each lane scans 8 pool columns of query-row l16 (half selects column group)
        const int pb = pbase + tile * 16 + half * 8;
#pragma unroll
        for (int t = 0; t < 8; ++t) {
            float v = sc[l16][half * 8 + t];
            int idx = pb + t;
            if (v > tv[KSEL - 1]) {  // strict '>' keeps lowest index on ties
                int pos = KSEL - 1;
                while (pos > 0 && v > tv[pos - 1]) {
                    tv[pos] = tv[pos - 1]; ti[pos] = ti[pos - 1]; --pos;
                }
                tv[pos] = v; ti[pos] = idx;
            }
        }
        __syncthreads();
    }

    // merge the two half-wave top-7 lists per query row, emit candidate indices
    __shared__ float mv[32][KSEL];
    __shared__ int   mi[32][KSEL];
#pragma unroll
    for (int j = 0; j < KSEL; ++j) { mv[lane][j] = tv[j]; mi[lane][j] = ti[j]; }
    __syncthreads();
    if (half == 0) {
        int ia = 0, ib = 0;
        int* out = g_top + ((size_t)(qtile * NSLICE + slice) * 16 + l16) * KSEL;
#pragma unroll
        for (int j = 0; j < KSEL; ++j) {
            float av = mv[l16][ia], bv = mv[l16 + 16][ib];
            bool ta = (av > bv) || (av == bv && mi[l16][ia] < mi[l16 + 16][ib]);
            if (ta) { out[j] = mi[l16][ia]; ++ia; }
            else    { out[j] = mi[l16 + 16][ib]; ++ib; }
        }
    }
}

// ---------------------------------------------------------------------------
// Per query: exact fp32 rescore of 350 candidates, exact top-7, GCN fusion,
// anomaly score. One 256-thread block per query. Precise expf here so the
// final ranking among candidates matches the reference.
// ---------------------------------------------------------------------------
__global__ __launch_bounds__(256) void fuse_kernel(const float* __restrict__ qe,
                                                   const float* __restrict__ qt,
                                                   const float* __restrict__ pe,
                                                   const float* __restrict__ pt,
                                                   const float* __restrict__ lam_p,
                                                   const float* __restrict__ W,
                                                   const float* __restrict__ bvec,
                                                   const float* __restrict__ pinv,
                                                   const int* __restrict__ g_top,
                                                   float* __restrict__ out) {
    const int n = blockIdx.x;
    const int tid = threadIdx.x;
    const int qtile = n >> 4, r = n & 15;
    const int NC = NSLICE * KSEL;  // 350

    __shared__ float qn_s[DIM];
    __shared__ int   cidx[NSLICE * KSEL];
    __shared__ float cscore[NSLICE * KSEL];
    __shared__ float red[256];
    __shared__ int   topi[KSEL];
    __shared__ float demos[KSEL][DIM];

    // normalized query in LDS
    float qd = qe[(size_t)n * DIM + tid];
    red[tid] = qd * qd;
    __syncthreads();
    for (int s = 128; s > 0; s >>= 1) { if (tid < s) red[tid] += red[tid + s]; __syncthreads(); }
    float qinv = 1.0f / fmaxf(sqrtf(red[0]), EPSF);
    qn_s[tid] = qd * qinv;
    __syncthreads();

    // gather candidate indices
    for (int c = tid; c < NC; c += 256) {
        int s = c / KSEL, j = c % KSEL;
        cidx[c] = g_top[((size_t)(qtile * NSLICE + s) * 16 + r) * KSEL + j];
    }
    __syncthreads();

    // exact fp32 rescore: one wave per candidate, lanes over D
    const float lam = lam_p[0];
    const float tq = qt[n];
    const int wave = tid >> 5, lane = tid & 31;
    for (int c = wave; c < NC; c += 8) {
        int p = cidx[c];
        const float* prow = pe + (size_t)p * DIM;
        float acc = 0.f;
#pragma unroll
        for (int u = 0; u < 8; ++u) acc += qn_s[lane + 32 * u] * prow[lane + 32 * u];
        for (int o = 16; o > 0; o >>= 1) acc += __shfl_xor(acc, o);
        if (lane == 0) {
            float tw = expf(-lam * fabsf(tq - pt[p]));
            cscore[c] = acc * pinv[p] * tw;
        }
    }
    __syncthreads();

    // exact top-7 (tiny, serial; lowest pool index wins ties like jax top_k)
    if (tid == 0) {
        for (int k = 0; k < KSEL; ++k) {
            float best = -FLT_MAX; int bi = 0; int bp = 0x7fffffff;
            for (int c = 0; c < NC; ++c) {
                float v = cscore[c]; int p = cidx[c];
                if (v > best || (v == best && p < bp)) { best = v; bi = c; bp = p; }
            }
            topi[k] = cidx[bi];
            cscore[bi] = -FLT_MAX;
        }
    }
    __syncthreads();

    // gather demos (raw pool rows)
#pragma unroll
    for (int k = 0; k < KSEL; ++k) demos[k][tid] = pe[(size_t)topi[k] * DIM + tid];
    __syncthreads();

    // h[k][tid] = sum_d demos[k][d] * W[d][tid]  (fp32, W column access coalesced)
    float acc[KSEL];
#pragma unroll
    for (int k = 0; k < KSEL; ++k) acc[k] = 0.f;
    for (int d = 0; d < DIM; ++d) {
        float w = W[(size_t)d * DIM + tid];
#pragma unroll
        for (int k = 0; k < KSEL; ++k) acc[k] += demos[k][d] * w;
    }
    // chain-GCN + mean over k folded: coef[k] = self_c[k] + (k<6 ? edge_c[k] : 0)
    // deg = [1,2,2,2,2,2,2]; self_c = [1,.5,...]; edge_c = [1/sqrt(2), .5 x5]
    const float coef[KSEL] = {1.70710678f, 1.f, 1.f, 1.f, 1.f, 1.f, 0.5f};
    float fsum = 0.f;
#pragma unroll
    for (int k = 0; k < KSEL; ++k) fsum += coef[k] * acc[k];
    float f = fsum * (1.0f / 7.0f) + bvec[tid];

    // reductions: ||fused||^2, qn.fused, ||q - fused||^2
    red[tid] = f * f;
    __syncthreads();
    for (int s = 128; s > 0; s >>= 1) { if (tid < s) red[tid] += red[tid + s]; __syncthreads(); }
    float sf = red[0];
    __syncthreads();
    red[tid] = qn_s[tid] * f;
    __syncthreads();
    for (int s = 128; s > 0; s >>= 1) { if (tid < s) red[tid] += red[tid + s]; __syncthreads(); }
    float qf = red[0];
    __syncthreads();
    red[tid] = (qd - f) * (qd - f);
    __syncthreads();
    for (int s = 128; s > 0; s >>= 1) { if (tid < s) red[tid] += red[tid + s]; __syncthreads(); }
    float dd = red[0];

    if (tid == 0) {
        float cosv = qf / fmaxf(sqrtf(sf), EPSF);
        out[n] = ALPHA * (1.0f - cosv) + BETA * sqrtf(dd);
    }
}

// ---------------------------------------------------------------------------
extern "C" void kernel_launch(void* const* d_in, const int* in_sizes, int n_in,
                              void* d_out, int out_size, void* d_ws, size_t ws_size,
                              hipStream_t stream) {
    (void)in_sizes; (void)n_in; (void)out_size; (void)ws_size;
    const float* qe  = (const float*)d_in[0];  // [N, D]
    const float* qt  = (const float*)d_in[1];  // [N]
    const float* pe  = (const float*)d_in[2];  // [M, D]
    const float* pt  = (const float*)d_in[3];  // [M]
    const float* lam = (const float*)d_in[4];  // scalar
    const float* W   = (const float*)d_in[5];  // [D, D]
    const float* b   = (const float*)d_in[6];  // [D]
    float* out = (float*)d_out;

    char* ws = (char*)d_ws;
    size_t o = 0;
    __bf16* pn = (__bf16*)(ws + o); o += (size_t)M_P * DIM * sizeof(__bf16);  // 51.2 MB
    float* pinv = (float*)(ws + o); o += (size_t)M_P * sizeof(float);         // 0.4 MB
    __bf16* qn = (__bf16*)(ws + o); o += (size_t)N_Q * DIM * sizeof(__bf16);  // 0.5 MB
    int* g_top = (int*)(ws + o);    o += (size_t)QTILES * NSLICE * 16 * KSEL * sizeof(int);

    hipLaunchKernelGGL(norm_rows_kernel, dim3(M_P), dim3(256), 0, stream, pe, pn, pinv);
    hipLaunchKernelGGL(norm_rows_kernel, dim3(N_Q), dim3(256), 0, stream, qe, qn, (float*)nullptr);
    hipLaunchKernelGGL(sim_topk_kernel, dim3(QTILES, NSLICE), dim3(32), 0, stream,
                       qn, pn, qt, pt, lam, g_top);
    hipLaunchKernelGGL(fuse_kernel, dim3(N_Q), dim3(256), 0, stream,
                       qe, qt, pe, pt, lam, W, b, pinv, g_top, out);
}
